// Injector_bimodal_32272384262572
// MI455X (gfx1250) — compile-verified, matrix operand order checked by reading, not run on hardware
//
#include <hip/hip_runtime.h>

typedef __attribute__((ext_vector_type(16))) _Float16 v16h;
typedef __attribute__((ext_vector_type(8)))  float    v8f;
typedef __attribute__((ext_vector_type(4)))  _Float16 h4;
typedef int ibm_v4i __attribute__((vector_size(16)));

#define LQn 9216
#define Cn  768
#define NHn 6
#define NPn 4
#define HWn 96

#define BM 128
#define BN 128
#define BK 32

#if defined(__HIP_DEVICE_COMPILE__) && __has_builtin(__builtin_amdgcn_global_load_async_to_lds_b128)
#define IBM_ASYNC 1
#else
#define IBM_ASYNC 0
#endif

// ---------------------------------------------------------------- convert f32 -> f16
__global__ void ibm_cvt_f16(const float* __restrict__ src, _Float16* __restrict__ dst, int n) {
    int i = blockIdx.x * blockDim.x + threadIdx.x;
    if (i < n) dst[i] = (_Float16)src[i];
}

// ---------------------------------------------------------------- LayerNorm (row = 768), f16 out
__global__ __launch_bounds__(256) void ibm_layernorm_f16(const float* __restrict__ x,
                                                         const float* __restrict__ w,
                                                         const float* __restrict__ b,
                                                         _Float16* __restrict__ y) {
    __shared__ float s1[256], s2[256];
    const int row = blockIdx.x;
    const float* xr = x + (size_t)row * Cn;
    float lx[3];
    float sum = 0.f, sq = 0.f;
#pragma unroll
    for (int j = 0; j < 3; ++j) {
        float v = xr[threadIdx.x + j * 256];
        lx[j] = v; sum += v; sq += v * v;
    }
    s1[threadIdx.x] = sum; s2[threadIdx.x] = sq;
    __syncthreads();
    for (int off = 128; off > 0; off >>= 1) {
        if (threadIdx.x < off) {
            s1[threadIdx.x] += s1[threadIdx.x + off];
            s2[threadIdx.x] += s2[threadIdx.x + off];
        }
        __syncthreads();
    }
    const float mu = s1[0] * (1.0f / Cn);
    const float var = s2[0] * (1.0f / Cn) - mu * mu;
    const float rs = rsqrtf(var + 1e-6f);
#pragma unroll
    for (int j = 0; j < 3; ++j) {
        int c = threadIdx.x + j * 256;
        y[(size_t)row * Cn + c] = (_Float16)((lx[j] - mu) * rs * w[c] + b[c]);
    }
}

// ---------------------------------------------------------------- async/sync 16B global->LDS copy
__device__ __forceinline__ void ibm_cp16(const _Float16* g, _Float16* l) {
#if IBM_ASYNC
    __builtin_amdgcn_global_load_async_to_lds_b128(
        (__attribute__((address_space(1))) ibm_v4i*)g,
        (__attribute__((address_space(3))) ibm_v4i*)l, 0, 0);
#else
    *(uint4*)l = *(const uint4*)g;
#endif
}

__device__ __forceinline__ void ibm_async_wait() {
#if IBM_ASYNC
#if __has_builtin(__builtin_amdgcn_s_wait_asynccnt)
    __builtin_amdgcn_s_wait_asynccnt(0);
#else
    asm volatile("s_wait_asynccnt 0x0" ::: "memory");
#endif
#endif
}

// ---------------------------------------------------------------- WMMA fragment loader
// 16-bit A/B 16x32 layout (ISA 7.12.2): lanes 0-15 hold row=lane, K pairs
//   VGPR v<4: K=2v,2v+1 ; v>=4: K=16+2(v-4) ; lanes 16-31: same rows, K+8.
__device__ __forceinline__ v16h ibm_frag_ld(const _Float16* rowbase, int half) {
    union { v16h v; unsigned u[8]; } f;
    const unsigned* p = (const unsigned*)rowbase;
#pragma unroll
    for (int v = 0; v < 8; ++v) {
        int k = (v < 4 ? 2 * v : 16 + 2 * (v - 4)) + half * 8;
        f.u[v] = p[k >> 1];
    }
    return f.v;
}

// ---------------------------------------------------------------- GEMM: D[M,N] = A[M,K] * W[N,K]^T + bias
// mode 0: outH (f16)   mode 1: outF (f32)   mode 2: outF = resid + gamma*(acc+bias)
__global__ __launch_bounds__(256) void ibm_gemm_wmma(const _Float16* __restrict__ A,
                                                     const _Float16* __restrict__ W,
                                                     const float* __restrict__ bias,
                                                     int M, int N, int K, int mode,
                                                     const float* __restrict__ resid,
                                                     const float* __restrict__ gamma,
                                                     float* __restrict__ outF,
                                                     _Float16* __restrict__ outH) {
    __shared__ _Float16 As[2][BM][BK];
    __shared__ _Float16 Bs[2][BN][BK];

    const int tid  = threadIdx.x;
    const int lane = tid & 31;
    const int wave = tid >> 5;     // 0..7
    const int wm   = wave >> 1;    // 0..3 -> 32-row slab
    const int wn   = wave & 1;     // 0..1 -> 64-col slab
    const int half = lane >> 4;
    const int lrow = lane & 15;
    const int m0 = blockIdx.y * BM;
    const int n0 = blockIdx.x * BN;

    const int rowL = tid >> 2;        // 0..63 loader row
    const int c8   = (tid & 3) << 3;  // 0,8,16,24 (half index)

    // zero-fill the B pad rows once (async copies can't zero-fill; N may be 48/24)
    if (N < BN) {
#pragma unroll
        for (int buf = 0; buf < 2; ++buf)
#pragma unroll
            for (int rr = 0; rr < BN; rr += 64)
                if (n0 + rowL + rr >= N)
                    *(uint4*)(&Bs[buf][rowL + rr][c8]) = make_uint4(0u, 0u, 0u, 0u);
    }

    v8f acc[2][4];
#pragma unroll
    for (int t = 0; t < 2; ++t)
#pragma unroll
        for (int j = 0; j < 4; ++j) acc[t][j] = {};

    // stage first K-slab into buffer 0
    {
#pragma unroll
        for (int rr = 0; rr < BM; rr += 64)
            ibm_cp16(A + (size_t)(m0 + rowL + rr) * K + c8, &As[0][rowL + rr][c8]);
#pragma unroll
        for (int rr = 0; rr < BN; rr += 64) {
            int n = n0 + rowL + rr;
            if (n < N) ibm_cp16(W + (size_t)n * K + c8, &Bs[0][rowL + rr][c8]);
        }
    }
    ibm_async_wait();
    __syncthreads();

    int cur = 0;
    for (int kk = 0; kk < K; kk += BK) {
        const int nxt = cur ^ 1;
        // stream next K-slab while computing this one
        if (kk + BK < K) {
            const int kn = kk + BK;
#pragma unroll
            for (int rr = 0; rr < BM; rr += 64)
                ibm_cp16(A + (size_t)(m0 + rowL + rr) * K + kn + c8, &As[nxt][rowL + rr][c8]);
#pragma unroll
            for (int rr = 0; rr < BN; rr += 64) {
                int n = n0 + rowL + rr;
                if (n < N) ibm_cp16(W + (size_t)n * K + kn + c8, &Bs[nxt][rowL + rr][c8]);
            }
            __builtin_prefetch(A + (size_t)(m0 + rowL) * K + kn + BK + c8, 0, 0);
        }

        v16h afr[2], bfr[4];
#pragma unroll
        for (int t = 0; t < 2; ++t) afr[t] = ibm_frag_ld(&As[cur][wm * 32 + t * 16 + lrow][0], half);
#pragma unroll
        for (int j = 0; j < 4; ++j) bfr[j] = ibm_frag_ld(&Bs[cur][wn * 64 + j * 16 + lrow][0], half);

#pragma unroll
        for (int t = 0; t < 2; ++t)
#pragma unroll
            for (int j = 0; j < 4; ++j)
                acc[t][j] = __builtin_amdgcn_wmma_f32_16x16x32_f16(
                    false, afr[t], false, bfr[j], (short)0, acc[t][j], false, false);

        ibm_async_wait();
        __syncthreads();
        cur = nxt;
    }

    // epilogue: D element r -> (m = r + 8*half, n = lrow) within each 16x16 tile
#pragma unroll
    for (int t = 0; t < 2; ++t) {
#pragma unroll
        for (int j = 0; j < 4; ++j) {
            int n = n0 + wn * 64 + j * 16 + lrow;
            if (n >= N) continue;
            float bv = bias ? bias[n] : 0.f;
#pragma unroll
            for (int r = 0; r < 8; ++r) {
                int m = m0 + wm * 32 + t * 16 + r + half * 8;
                float v = acc[t][j][r] + bv;
                if (mode == 0)      outH[(size_t)m * N + n] = (_Float16)v;
                else if (mode == 1) outF[(size_t)m * N + n] = v;
                else                outF[(size_t)m * N + n] = resid[(size_t)m * N + n] + gamma[n] * v;
            }
        }
    }
}

// ---------------------------------------------------------------- MSDeform sampling core
// thread = (lq, head, 4 channels). softmax over 4 points, bilinear gather (zeros padding).
__global__ __launch_bounds__(256) void ibm_msdeform_sample(const _Float16* __restrict__ value, // [LQ, C]
                                                           const float* __restrict__ ref,      // [LQ, 2]
                                                           const float* __restrict__ so,       // [LQ, 48]
                                                           const float* __restrict__ aw,       // [LQ, 24]
                                                           _Float16* __restrict__ out) {       // [LQ, C]
    int tid = blockIdx.x * blockDim.x + threadIdx.x;
    int dgrp = tid & 31;
    int head = (tid >> 5) % NHn;
    int lq = tid / (32 * NHn);
    if (lq >= LQn) return;
    int d0 = head * 128 + dgrp * 4;

    float rx = ref[lq * 2 + 0] * HWn - 0.5f;
    float ry = ref[lq * 2 + 1] * HWn - 0.5f;

    float lg[NPn];
    float mx = -1e30f;
#pragma unroll
    for (int p = 0; p < NPn; ++p) { lg[p] = aw[lq * 24 + head * 4 + p]; mx = fmaxf(mx, lg[p]); }
    float den = 0.f;
#pragma unroll
    for (int p = 0; p < NPn; ++p) { lg[p] = __expf(lg[p] - mx); den += lg[p]; }
    float rden = 1.f / den;

    float a0 = 0.f, a1 = 0.f, a2 = 0.f, a3 = 0.f;
#pragma unroll
    for (int p = 0; p < NPn; ++p) {
        float px = rx + so[lq * 48 + head * 8 + p * 2 + 0];
        float py = ry + so[lq * 48 + head * 8 + p * 2 + 1];
        float fx0 = floorf(px), fy0 = floorf(py);
        int x0 = (int)fx0, y0 = (int)fy0;
        float fx = px - fx0, fy = py - fy0;
        float wgt = lg[p] * rden;
        float w00 = (1.f - fy) * (1.f - fx) * wgt;
        float w01 = (1.f - fy) * fx * wgt;
        float w10 = fy * (1.f - fx) * wgt;
        float w11 = fy * fx * wgt;
#pragma unroll
        for (int cy = 0; cy < 2; ++cy) {
            int yy = y0 + cy;
            if ((unsigned)yy >= (unsigned)HWn) continue;
#pragma unroll
            for (int cx = 0; cx < 2; ++cx) {
                int xx = x0 + cx;
                if ((unsigned)xx >= (unsigned)HWn) continue;
                float w = cy ? (cx ? w11 : w10) : (cx ? w01 : w00);
                h4 v = *(const h4*)(value + (size_t)(yy * HWn + xx) * Cn + d0);
                a0 += w * (float)v.x; a1 += w * (float)v.y;
                a2 += w * (float)v.z; a3 += w * (float)v.w;
            }
        }
    }
    h4 o;
    o.x = (_Float16)a0; o.y = (_Float16)a1; o.z = (_Float16)a2; o.w = (_Float16)a3;
    *(h4*)(out + (size_t)lq * Cn + d0) = o;
}

// ---------------------------------------------------------------- host side
namespace {
constexpr size_t HALF_ACT  = (size_t)LQn * Cn * 2;   // 14,155,776 B
constexpr size_t OFF_QN    = 0;
constexpr size_t OFF_FN    = OFF_QN + HALF_ACT;
constexpr size_t OFF_VAL   = OFF_FN + HALF_ACT;
constexpr size_t OFF_ATTN  = OFF_VAL + HALF_ACT;
constexpr size_t OFF_SO    = OFF_ATTN + HALF_ACT;                 // f32 [LQ,48]
constexpr size_t OFF_AW    = OFF_SO + (size_t)LQn * 48 * 4;       // f32 [LQ,24]
constexpr size_t OFF_Q     = OFF_AW + (size_t)LQn * 24 * 4;       // f32 [LQ,C]
constexpr size_t OFF_WSO   = OFF_Q + (size_t)LQn * Cn * 4;        // f16 48x768
constexpr size_t OFF_WAW   = OFF_WSO + (size_t)48 * Cn * 2;       // f16 24x768
constexpr size_t OFF_WVP   = OFF_WAW + (size_t)24 * Cn * 2;       // f16 768x768
constexpr size_t OFF_WOP   = OFF_WVP + (size_t)Cn * Cn * 2;       // f16 768x768

struct PassW {
    const float *qn_w, *qn_b, *fn_w, *fn_b;
    const float *so_w, *so_b, *aw_w, *aw_b;
    const float *vp_w, *vp_b, *op_w, *op_b, *gamma;
};

void run_pass(const float* q_in, const float* feat_in, const float* ref,
              const PassW& pw, char* ws, float* q_out, hipStream_t stream) {
    _Float16* qn_h   = (_Float16*)(ws + OFF_QN);
    _Float16* fn_h   = (_Float16*)(ws + OFF_FN);
    _Float16* val_h  = (_Float16*)(ws + OFF_VAL);
    _Float16* attn_h = (_Float16*)(ws + OFF_ATTN);
    float*    so_o   = (float*)(ws + OFF_SO);
    float*    aw_o   = (float*)(ws + OFF_AW);
    _Float16* wso_h  = (_Float16*)(ws + OFF_WSO);
    _Float16* waw_h  = (_Float16*)(ws + OFF_WAW);
    _Float16* wvp_h  = (_Float16*)(ws + OFF_WVP);
    _Float16* wop_h  = (_Float16*)(ws + OFF_WOP);

    // weight conversion
    ibm_cvt_f16<<<(48 * Cn + 255) / 256, 256, 0, stream>>>(pw.so_w, wso_h, 48 * Cn);
    ibm_cvt_f16<<<(24 * Cn + 255) / 256, 256, 0, stream>>>(pw.aw_w, waw_h, 24 * Cn);
    ibm_cvt_f16<<<(Cn * Cn + 255) / 256, 256, 0, stream>>>(pw.vp_w, wvp_h, Cn * Cn);
    ibm_cvt_f16<<<(Cn * Cn + 255) / 256, 256, 0, stream>>>(pw.op_w, wop_h, Cn * Cn);

    // layernorms -> f16 GEMM operands
    ibm_layernorm_f16<<<LQn, 256, 0, stream>>>(q_in, pw.qn_w, pw.qn_b, qn_h);
    ibm_layernorm_f16<<<LQn, 256, 0, stream>>>(feat_in, pw.fn_w, pw.fn_b, fn_h);

    dim3 blk(256);
    // value projection  [LQ,768] = fn * vp^T + b  -> f16
    ibm_gemm_wmma<<<dim3(Cn / BN, LQn / BM), blk, 0, stream>>>(
        fn_h, wvp_h, pw.vp_b, LQn, Cn, Cn, 0, nullptr, nullptr, nullptr, val_h);
    // sampling offsets  [LQ,48] -> f32
    ibm_gemm_wmma<<<dim3(1, LQn / BM), blk, 0, stream>>>(
        qn_h, wso_h, pw.so_b, LQn, 48, Cn, 1, nullptr, nullptr, so_o, nullptr);
    // attention logits  [LQ,24] -> f32
    ibm_gemm_wmma<<<dim3(1, LQn / BM), blk, 0, stream>>>(
        qn_h, waw_h, pw.aw_b, LQn, 24, Cn, 1, nullptr, nullptr, aw_o, nullptr);

    // deformable sampling -> attn f16
    ibm_msdeform_sample<<<(LQn * NHn * 32) / 256, 256, 0, stream>>>(val_h, ref, so_o, aw_o, attn_h);

    // output projection + residual:  q_out = q_in + gamma * (attn * op^T + b)
    ibm_gemm_wmma<<<dim3(Cn / BN, LQn / BM), blk, 0, stream>>>(
        attn_h, wop_h, pw.op_b, LQn, Cn, Cn, 2, q_in, pw.gamma, q_out, nullptr);
}
} // namespace

extern "C" void kernel_launch(void* const* d_in, const int* in_sizes, int n_in,
                              void* d_out, int out_size, void* d_ws, size_t ws_size,
                              hipStream_t stream) {
    const float* query = (const float*)d_in[0];
    const float* ref   = (const float*)d_in[1];
    const float* feat  = (const float*)d_in[2];
    const float* feat2 = (const float*)d_in[3];
    // d_in[4] spatial_shapes (=96x96), d_in[5] level_start_index: compile-time constants here.

    PassW p1 { (const float*)d_in[6],  (const float*)d_in[7],
               (const float*)d_in[8],  (const float*)d_in[9],
               (const float*)d_in[10], (const float*)d_in[11],
               (const float*)d_in[12], (const float*)d_in[13],
               (const float*)d_in[14], (const float*)d_in[15],
               (const float*)d_in[16], (const float*)d_in[17],
               (const float*)d_in[18] };
    PassW p2 { (const float*)d_in[19], (const float*)d_in[20],
               (const float*)d_in[21], (const float*)d_in[22],
               (const float*)d_in[23], (const float*)d_in[24],
               (const float*)d_in[25], (const float*)d_in[26],
               (const float*)d_in[27], (const float*)d_in[28],
               (const float*)d_in[29], (const float*)d_in[30],
               (const float*)d_in[31] };

    char* ws = (char*)d_ws;
    float* q_mid = (float*)(ws + OFF_Q);

    // pass 1: q_mid = query + gamma * MSDeformAttn(LN(query), LN(feat))
    run_pass(query, feat, ref, p1, ws, q_mid, stream);
    // pass 2: out = q_mid + gamma2 * MSDeformAttn(LN(q_mid), LN(feat_other))
    run_pass(q_mid, feat2, ref, p2, ws, (float*)d_out, stream);
}
